// GCRN_33741263077849
// MI455X (gfx1250) — matrix operand order repaired and checked by reading, not compiled, not run on hardware
//
#include <hip/hip_runtime.h>
#include <hip/hip_bf16.h>
#include <math.h>

// ---------------- types / WMMA helpers (gfx1250 wave32) ----------------
typedef _Float16 h16;
typedef __attribute__((ext_vector_type(16))) _Float16 v16h;
typedef __attribute__((ext_vector_type(8)))  _Float16 v8h;
typedef __attribute__((ext_vector_type(8)))  float    v8f;

#define WMMA_F16(a, b, c) \
  __builtin_amdgcn_wmma_f32_16x16x32_f16(false, (a), false, (b), (short)0, (c), false, false)

union FragH  { v16h v; h16  a[16]; };
union Frag8H { v8h  v; h16  a[8];  };
union Acc    { v8f  v; float f[8]; };

__device__ __forceinline__ void zacc(Acc& a) {
#pragma unroll
  for (int r = 0; r < 8; ++r) a.f[r] = 0.f;
}

// A-matrix fragment (16x32 f16): lane -> row = lane&15; element e -> k = (e&7) + (e>>3)*16 + hi*8
__device__ __forceinline__ v16h load_a_frag(const h16* base, int ld, int lo, int hi, int kbase) {
  const h16* p = base + (size_t)lo * ld + kbase + hi * 8;
  FragH u;
  *(v8h*)&u.a[0] = *(const v8h*)(p);
  *(v8h*)&u.a[8] = *(const v8h*)(p + 16);
  return u.v;
}

// B-matrix fragment (32x16 f16), source stored K-contiguous per column: src[col][k], ld = K stride
// lane -> col = lane&15; element e -> k = (e&7) + (e>>3)*16 + hi*8
__device__ __forceinline__ v16h load_b_frag(const h16* base, int ld, int lo, int hi, int kbase, int col) {
  const h16* p = base + (size_t)(col + lo) * ld + kbase + hi * 8;
  FragH u;
  *(v8h*)&u.a[0] = *(const v8h*)(p);
  *(v8h*)&u.a[8] = *(const v8h*)(p + 16);
  return u.v;
}

// Async-stage `nrows` rows of `rowElems` contiguous f16 (row stride gStrideElems) into LDS.
// Uses gfx1250 GLOBAL_LOAD_ASYNC_TO_LDS_B128 (ASYNCcnt); caller must async_wait()+__syncthreads().
__device__ __forceinline__ void stage_rows_async(h16* lds_dst, const h16* gsrc,
                                                 int nrows, int rowElems, int gStrideElems, int t) {
  const int upr   = rowElems >> 3;          // 16B units per row
  const int units = nrows * upr;
  const uint32_t lbase = (uint32_t)(uintptr_t)lds_dst;
  for (int u = t; u < units; u += 256) {
    const int g = u / upr, cu = u - g * upr;
    const uint32_t laddr = lbase + (uint32_t)u * 16u;
    const uint64_t ga = (uint64_t)(uintptr_t)(gsrc + (size_t)g * gStrideElems + cu * 8);
    asm volatile("global_load_async_to_lds_b128 %0, %1, off"
                 :: "v"(laddr), "v"(ga) : "memory");
  }
}
__device__ __forceinline__ void async_wait() {
  asm volatile("s_wait_asynccnt 0x0" ::: "memory");
}

// ---------------- problem constants ----------------
constexpr int   cB = 8, cN = 512, cF = 64, cG = 64, cM = 4, cE = 4, cL0 = 256;
constexpr int   BEM = cB * cE * cM;        // 128
constexpr int   RROWS = cB * cN;           // 4096
constexpr int   K1D = cM * cE * cG;        // 1024
constexpr float ALPHA = 0.5f, EPS = 1e-5f, SLOPE = 0.01f;

// ---------------- workspace offsets (bytes) ----------------
constexpr size_t OFF_WHT   = 0;                                       // [128][64][512] f16
constexpr size_t OFF_S1    = OFF_WHT   + (size_t)BEM * cG * cN * 2;
constexpr size_t OFF_S2    = OFF_S1    + (size_t)BEM * cN * 4;
constexpr size_t OFF_HCAT  = OFF_S2    + (size_t)BEM * cN * 4;        // [4096][1024] f16
constexpr size_t OFF_HMID  = OFF_HCAT  + (size_t)RROWS * K1D * 2;     // [4096][64] f32
constexpr size_t OFF_H1    = OFF_HMID  + (size_t)RROWS * cF * 4;      // [4096][64] f32
constexpr size_t OFF_H1H   = OFF_H1    + (size_t)RROWS * cF * 4;      // [4096][64] f16
constexpr size_t OFF_T     = OFF_H1H   + (size_t)RROWS * cF * 2;      // [4096][256] f32
constexpr size_t OFF_T2H   = OFF_T     + (size_t)RROWS * cL0 * 4;     // [4096][256] f16
constexpr size_t OFF_ZP    = OFF_T2H   + (size_t)RROWS * cL0 * 2;     // [4096][64] f32
constexpr size_t OFF_WE1H  = OFF_ZP    + (size_t)RROWS * cF * 4;      // [64][1024] f16
constexpr size_t OFF_W2AH  = OFF_WE1H  + (size_t)cF * K1D * 2;        // [256][64] f16
constexpr size_t OFF_W2BH  = OFF_W2AH  + (size_t)cL0 * cF * 2;        // [64][256] f16
constexpr size_t OFF_STATS = OFF_W2BH  + (size_t)cF * cL0 * 2;        // 768 floats

// =============== K0: convert weights to f16, zero BN stats ===============
__global__ __launch_bounds__(256) void k0_prep(const float* __restrict__ We1,
                                               const float* __restrict__ W2a,
                                               const float* __restrict__ W2b,
                                               h16* We1h, h16* W2ah, h16* W2bh,
                                               float* stats) {
  int t = blockIdx.x * 256 + threadIdx.x;
  if (t < 65536)       We1h[t]         = (h16)We1[t];
  else if (t < 81920)  W2ah[t - 65536] = (h16)W2a[t - 65536];
  else if (t < 98304)  W2bh[t - 81920] = (h16)W2b[t - 81920];
  if (t < 768) stats[t] = 0.f;
}

// =============== K1: Wh = X @ W  (per b,e,m), store Wh^T f16 [g][n] ===============
__global__ __launch_bounds__(256) void k1_proj(const float* __restrict__ X,
                                               const float* __restrict__ Ws,
                                               h16* __restrict__ Wht) {
  __shared__ h16 Wt[64 * 64];      // W transposed: Wt[g][f] -> K-contiguous B source
  __shared__ h16 Xs[128 * 64];     // 128-row X chunk, f16
  const int bem = blockIdx.x;
  const int b = bem >> 4, e = (bem >> 2) & 3, m = bem & 3;
  const int t = threadIdx.x, wv = t >> 5, lane = t & 31, lo = lane & 15, hi = lane >> 4;

  const float* Wsrc = Ws + (size_t)(e * 256 + m * 64) * 64;   // [f][g]
  for (int i = t; i < 4096; i += 256) { int f = i >> 6, g = i & 63; Wt[g * 64 + f] = (h16)Wsrc[i]; }

  h16* Wout = Wht + (size_t)bem * (cG * cN);
  for (int chunk = 0; chunk < 4; ++chunk) {
    __syncthreads();
    const float* Xsrc = X + (size_t)(b * cN + chunk * 128) * cF;
    for (int i = t; i < 8192; i += 256) Xs[i] = (h16)Xsrc[i];
    __syncthreads();

    const int l0 = wv * 16;
#pragma unroll
    for (int c = 0; c < 4; ++c) {
      Acc acc; zacc(acc);
#pragma unroll
      for (int kt = 0; kt < 2; ++kt) {
        v16h af = load_a_frag(Xs + (size_t)l0 * 64, 64, lo, hi, kt * 32);
        v16h bf = load_b_frag(Wt, 64, lo, hi, kt * 32, c * 16);
        acc.v = WMMA_F16(af, bf, acc.v);
      }
      Frag8H hv;
#pragma unroll
      for (int r = 0; r < 8; ++r) hv.a[r] = (h16)acc.f[r];
      const int n0 = chunk * 128 + wv * 16 + hi * 8;
      *(v8h*)(Wout + (size_t)(c * 16 + lo) * cN + n0) = hv.v;
    }
  }
}

// =============== K1b: s1 = X·(W a1), s2 = X·(W a2)  (exact f32) ===============
__global__ __launch_bounds__(256) void k1b_scores(const float* __restrict__ X,
                                                  const float* __restrict__ Ws,
                                                  const float* __restrict__ a1,
                                                  const float* __restrict__ a2,
                                                  float* __restrict__ s1g,
                                                  float* __restrict__ s2g) {
  __shared__ float wa1[64], wa2[64];
  const int bem = blockIdx.x;
  const int b = bem >> 4, e = (bem >> 2) & 3, m = bem & 3;
  const int t = threadIdx.x;
  const float* Wsrc = Ws + (size_t)(e * 256 + m * 64) * 64;
  const float* a1v = a1 + e * 256 + m * 64;
  const float* a2v = a2 + e * 256 + m * 64;
  if (t < 128) {
    const int f = t & 63;
    const float* av = (t < 64) ? a1v : a2v;
    float s = 0.f;
    for (int g = 0; g < 64; ++g) s += Wsrc[f * 64 + g] * av[g];
    if (t < 64) wa1[f] = s; else wa2[f] = s;
  }
  __syncthreads();
  for (int n = t; n < cN; n += 256) {
    const float* xr = X + (size_t)(b * cN + n) * cF;
    float s1 = 0.f, s2 = 0.f;
    for (int f = 0; f < 64; ++f) { float x = xr[f]; s1 += x * wa1[f]; s2 += x * wa2[f]; }
    s1g[bem * cN + n] = s1;
    s2g[bem * cN + n] = s2;
  }
}

// =============== K2: single-pass masked softmax + att@Wh + ELU -> Hcat f16 ===============
// m̂ = leakyrelu(s1_i + max_j s2_j) upper-bounds every logit (leaky-relu monotone), so
// exp(l - m̂) <= 1 and softmax is shift-invariant: normalize by post-scaling the f32 acc.
__global__ __launch_bounds__(256) void k2_attn(const int* __restrict__ A,
                                               const float* __restrict__ s1g,
                                               const float* __restrict__ s2g,
                                               const h16* __restrict__ Wht,
                                               h16* __restrict__ Hcat) {
  __shared__ h16   WhL[64 * 256];          // staged half of Wh^T (j-half), 32KB
  __shared__ float s1s[128], s2s[512], rinvs[128], red[256];
  const int bid = blockIdx.x, bem = bid >> 2, chunk = bid & 3;
  const int b = bem >> 4, e = (bem >> 2) & 3, m = bem & 3;
  const int t = threadIdx.x, wv = t >> 5, lane = t & 31, lo = lane & 15, hi = lane >> 4;

  if (t < 128) s1s[t] = s1g[bem * cN + chunk * 128 + t];
  {
    float v0 = s2g[bem * cN + t], v1 = s2g[bem * cN + 256 + t];
    s2s[t] = v0; s2s[256 + t] = v1;
    red[t] = fmaxf(v0, v1);
  }
  __syncthreads();
  for (int s = 128; s > 0; s >>= 1) {
    if (t < s) red[t] = fmaxf(red[t], red[t + s]);
    __syncthreads();
  }
  const float s2max = red[0];

  const int iloc = wv * 16 + lo;                  // row this lane owns (A-frag row == lane&15)
  const int iabs = chunk * 128 + iloc;
  const int* Arow = A + ((size_t)((b * 4 + e) * cN + iabs)) * cN;
  const float s1v = s1s[iloc];
  float mhat = s1v + s2max;
  mhat = mhat > 0.f ? mhat : SLOPE * mhat;

  const h16* Wb = Wht + (size_t)bem * (cG * cN);
  Acc acc[4];
#pragma unroll
  for (int c = 0; c < 4; ++c) zacc(acc[c]);
  float ssum = 0.f;

  for (int h = 0; h < 2; ++h) {                   // j-halves of 256
    if (h) __syncthreads();                       // consumers done with previous stage
    stage_rows_async(WhL, Wb + h * 256, 64, 256, cN, t);
    async_wait();
    __syncthreads();

    for (int jt = 0; jt < 8; ++jt) {              // K = 32 per step
      const int jb = h * 256 + jt * 32 + hi * 8;
      int4 m0 = *(const int4*)(Arow + jb);
      int4 m1 = *(const int4*)(Arow + jb + 4);
      int4 m2 = *(const int4*)(Arow + jb + 16);
      int4 m3 = *(const int4*)(Arow + jb + 20);
      int mk[16] = {m0.x, m0.y, m0.z, m0.w, m1.x, m1.y, m1.z, m1.w,
                    m2.x, m2.y, m2.z, m2.w, m3.x, m3.y, m3.z, m3.w};
      FragH af;
#pragma unroll
      for (int ei = 0; ei < 16; ++ei) {
        const int j = jb + ((ei < 8) ? ei : (ei + 8));   // elements 8..15 -> k offset +16
        float l = s1v + s2s[j];
        l = l > 0.f ? l : SLOPE * l;
        float p = (mk[ei] > 0) ? expf(l - mhat) : 0.f;   // unnormalized prob, <= 1
        ssum += p;
        af.a[ei] = (h16)p;
      }
#pragma unroll
      for (int c = 0; c < 4; ++c) {
        v16h bf = load_b_frag(WhL, 256, lo, hi, jt * 32, c * 16);
        acc[c].v = WMMA_F16(af.v, bf, acc[c].v);
      }
    }
  }

  ssum += __shfl_xor(ssum, 16, 32);               // lane pair covers all 512 j of row `lo`
  if (hi == 0) rinvs[wv * 16 + lo] = 1.0f / fmaxf(ssum, 1e-30f);
  __syncthreads();

  const int nbase = chunk * 128 + wv * 16 + hi * 8;
#pragma unroll
  for (int c = 0; c < 4; ++c) {
#pragma unroll
    for (int r = 0; r < 8; ++r) {
      float hv = acc[c].f[r] * rinvs[wv * 16 + r + hi * 8];
      hv = hv > 0.f ? hv : expm1f(hv);
      const int n = nbase + r;
      const size_t idx = (size_t)(b * cN + n) * K1D + m * 256 + e * 64 + c * 16 + lo;
      Hcat[idx] = (h16)hv;
    }
  }
}

// =============== K3: Hmid = 0.5*(Hcat @ We1^T) + 0.5*X  (+ BN1 stats) ===============
__global__ __launch_bounds__(256) void k3_emb(const h16* __restrict__ Hcat,
                                              const h16* __restrict__ We1h,
                                              const float* __restrict__ X,
                                              float* __restrict__ Hmid,
                                              float* __restrict__ stats) {
  __shared__ h16   WeL[64 * 256];   // staged K-slice of W_emb1, 32KB
  __shared__ float ssum[64], ssq[64];
  const int t = threadIdx.x, wv = t >> 5, lane = t & 31, lo = lane & 15, hi = lane >> 4;
  if (t < 64) { ssum[t] = 0.f; ssq[t] = 0.f; }

  const int r0 = (blockIdx.x * 8 + wv) * 16;
  Acc acc[4];
#pragma unroll
  for (int c = 0; c < 4; ++c) zacc(acc[c]);

  for (int kg = 0; kg < 4; ++kg) {               // K groups of 256
    if (kg) __syncthreads();
    stage_rows_async(WeL, We1h + kg * 256, 64, 256, K1D, t);
    async_wait();
    __syncthreads();
#pragma unroll
    for (int kt2 = 0; kt2 < 8; ++kt2) {
      const int kt = kg * 8 + kt2;
      v16h af = load_a_frag(Hcat + (size_t)r0 * K1D, K1D, lo, hi, kt * 32);
#pragma unroll
      for (int c = 0; c < 4; ++c)
        acc[c].v = WMMA_F16(af, load_b_frag(WeL, 256, lo, hi, kt2 * 32, c * 16), acc[c].v);
    }
  }
#pragma unroll
  for (int c = 0; c < 4; ++c) {
#pragma unroll
    for (int r = 0; r < 8; ++r) {
      const int row = r0 + r + hi * 8, f = c * 16 + lo;
      const float v = (1.0f - ALPHA) * acc[c].f[r] + ALPHA * X[(size_t)row * cF + f];
      Hmid[(size_t)row * cF + f] = v;
      atomicAdd(&ssum[f], v);
      atomicAdd(&ssq[f], v * v);
    }
  }
  __syncthreads();
  if (t < 64) { atomicAdd(&stats[t], ssum[t]); atomicAdd(&stats[64 + t], ssq[t]); }
}

// =============== K4: BN1 apply -> H1 f32 + f16 ===============
__global__ __launch_bounds__(256) void k4_bn1(const float* __restrict__ Hmid,
                                              const float* __restrict__ stats,
                                              float* __restrict__ H1,
                                              h16* __restrict__ H1h) {
  __shared__ float mu[64], inv[64];
  const int t = threadIdx.x;
  if (t < 64) {
    const float s = stats[t] * (1.0f / RROWS);
    const float q = stats[64 + t] * (1.0f / RROWS) - s * s;
    mu[t] = s; inv[t] = rsqrtf(q + EPS);
  }
  __syncthreads();
  const int base = blockIdx.x * 1024 + t * 4;
#pragma unroll
  for (int q = 0; q < 4; ++q) {
    const int i = base + q, f = i & 63;
    const float v = (Hmid[i] - mu[f]) * inv[f];
    H1[i] = v; H1h[i] = (h16)v;
  }
}

// =============== K5: T = H1 @ W2a^T  [4096,256] (+ BN2 stats) ===============
__global__ __launch_bounds__(256) void k5_g2(const h16* __restrict__ H1h,
                                             const h16* __restrict__ W2ah,
                                             float* __restrict__ T,
                                             float* __restrict__ stats) {
  __shared__ h16   WaL[256 * 64];   // whole W2a f16, 32KB
  __shared__ float ssum[256], ssq[256];
  const int t = threadIdx.x, wv = t >> 5, lane = t & 31, lo = lane & 15, hi = lane >> 4;
  ssum[t] = 0.f; ssq[t] = 0.f;
  stage_rows_async(WaL, W2ah, 256, 64, 64, t);
  async_wait();
  __syncthreads();

  const int gw = blockIdx.x * 8 + wv;          // 0..511
  const int r0 = (gw >> 1) * 16, ch = gw & 1;  // 16 rows x 128 cols per wave
  Acc acc[8];
#pragma unroll
  for (int c = 0; c < 8; ++c) zacc(acc[c]);
#pragma unroll
  for (int kt = 0; kt < 2; ++kt) {
    v16h af = load_a_frag(H1h + (size_t)r0 * cF, cF, lo, hi, kt * 32);
#pragma unroll
    for (int c = 0; c < 8; ++c)
      acc[c].v = WMMA_F16(af, load_b_frag(WaL, cF, lo, hi, kt * 32, ch * 128 + c * 16), acc[c].v);
  }
#pragma unroll
  for (int c = 0; c < 8; ++c) {
#pragma unroll
    for (int r = 0; r < 8; ++r) {
      const int row = r0 + r + hi * 8, col = ch * 128 + c * 16 + lo;
      const float v = acc[c].f[r];
      T[(size_t)row * cL0 + col] = v;
      atomicAdd(&ssum[col], v);
      atomicAdd(&ssq[col], v * v);
    }
  }
  __syncthreads();
  atomicAdd(&stats[128 + t], ssum[t]);
  atomicAdd(&stats[384 + t], ssq[t]);
}

// =============== K6: BN2 + ELU -> f16 ===============
__global__ __launch_bounds__(256) void k6_bn2(const float* __restrict__ T,
                                              const float* __restrict__ stats,
                                              h16* __restrict__ T2h) {
  __shared__ float mu[256], inv[256];
  const int t = threadIdx.x;
  {
    const float s = stats[128 + t] * (1.0f / RROWS);
    const float q = stats[384 + t] * (1.0f / RROWS) - s * s;
    mu[t] = s; inv[t] = rsqrtf(q + EPS);
  }
  __syncthreads();
  const int base = blockIdx.x * 1024 + t * 4;
#pragma unroll
  for (int q = 0; q < 4; ++q) {
    const int i = base + q, f = i & 255;
    float v = (T[i] - mu[f]) * inv[f];
    v = v > 0.f ? v : expm1f(v);
    T2h[i] = (h16)v;
  }
}

// =============== K7: Zp = T2 @ W2b^T + H1  (+ BN3 stats) ===============
__global__ __launch_bounds__(256) void k7_g3(const h16* __restrict__ T2h,
                                             const h16* __restrict__ W2bh,
                                             const float* __restrict__ H1,
                                             float* __restrict__ Zp,
                                             float* __restrict__ stats) {
  __shared__ h16   WbL[64 * 256];   // whole W2b f16, 32KB
  __shared__ float ssum[64], ssq[64];
  const int t = threadIdx.x, wv = t >> 5, lane = t & 31, lo = lane & 15, hi = lane >> 4;
  if (t < 64) { ssum[t] = 0.f; ssq[t] = 0.f; }
  stage_rows_async(WbL, W2bh, 64, 256, 256, t);
  async_wait();
  __syncthreads();

  const int r0 = (blockIdx.x * 8 + wv) * 16;
  Acc acc[4];
#pragma unroll
  for (int c = 0; c < 4; ++c) zacc(acc[c]);
#pragma unroll
  for (int kt = 0; kt < 8; ++kt) {
    v16h af = load_a_frag(T2h + (size_t)r0 * cL0, cL0, lo, hi, kt * 32);
#pragma unroll
    for (int c = 0; c < 4; ++c)
      acc[c].v = WMMA_F16(af, load_b_frag(WbL, cL0, lo, hi, kt * 32, c * 16), acc[c].v);
  }
#pragma unroll
  for (int c = 0; c < 4; ++c) {
#pragma unroll
    for (int r = 0; r < 8; ++r) {
      const int row = r0 + r + hi * 8, f = c * 16 + lo;
      const float v = acc[c].f[r] + H1[(size_t)row * cF + f];
      Zp[(size_t)row * cF + f] = v;
      atomicAdd(&ssum[f], v);
      atomicAdd(&ssq[f], v * v);
    }
  }
  __syncthreads();
  if (t < 64) { atomicAdd(&stats[640 + t], ssum[t]); atomicAdd(&stats[704 + t], ssq[t]); }
}

// =============== K8: BN3 apply -> output f32 ===============
__global__ __launch_bounds__(256) void k8_bn3(const float* __restrict__ Zp,
                                              const float* __restrict__ stats,
                                              float* __restrict__ out) {
  __shared__ float mu[64], inv[64];
  const int t = threadIdx.x;
  if (t < 64) {
    const float s = stats[640 + t] * (1.0f / RROWS);
    const float q = stats[704 + t] * (1.0f / RROWS) - s * s;
    mu[t] = s; inv[t] = rsqrtf(q + EPS);
  }
  __syncthreads();
  const int base = blockIdx.x * 1024 + t * 4;
#pragma unroll
  for (int q = 0; q < 4; ++q) {
    const int i = base + q, f = i & 63;
    out[i] = (Zp[i] - mu[f]) * inv[f];
  }
}

// ---------------- launcher ----------------
extern "C" void kernel_launch(void* const* d_in, const int* in_sizes, int n_in,
                              void* d_out, int out_size, void* d_ws, size_t ws_size,
                              hipStream_t stream) {
  (void)in_sizes; (void)n_in; (void)out_size; (void)ws_size;
  const int*   A   = (const int*)d_in[0];
  const float* X   = (const float*)d_in[1];
  const float* Ws  = (const float*)d_in[2];
  const float* a1  = (const float*)d_in[3];
  const float* a2  = (const float*)d_in[4];
  const float* We1 = (const float*)d_in[5];
  const float* W2a = (const float*)d_in[6];
  const float* W2b = (const float*)d_in[7];
  float* out = (float*)d_out;

  char* ws = (char*)d_ws;
  h16*   Wht   = (h16*)(ws + OFF_WHT);
  float* s1g   = (float*)(ws + OFF_S1);
  float* s2g   = (float*)(ws + OFF_S2);
  h16*   Hcat  = (h16*)(ws + OFF_HCAT);
  float* Hmid  = (float*)(ws + OFF_HMID);
  float* H1    = (float*)(ws + OFF_H1);
  h16*   H1h   = (h16*)(ws + OFF_H1H);
  float* T     = (float*)(ws + OFF_T);
  h16*   T2h   = (h16*)(ws + OFF_T2H);
  float* Zp    = (float*)(ws + OFF_ZP);
  h16*   We1h  = (h16*)(ws + OFF_WE1H);
  h16*   W2ah  = (h16*)(ws + OFF_W2AH);
  h16*   W2bh  = (h16*)(ws + OFF_W2BH);
  float* stats = (float*)(ws + OFF_STATS);

  k0_prep   <<<384, 256, 0, stream>>>(We1, W2a, W2b, We1h, W2ah, W2bh, stats);
  k1_proj   <<<BEM, 256, 0, stream>>>(X, Ws, Wht);
  k1b_scores<<<BEM, 256, 0, stream>>>(X, Ws, a1, a2, s1g, s2g);
  k2_attn   <<<BEM * 4, 256, 0, stream>>>(A, s1g, s2g, Wht, Hcat);
  k3_emb    <<<32, 256, 0, stream>>>(Hcat, We1h, X, Hmid, stats);
  k4_bn1    <<<256, 256, 0, stream>>>(Hmid, stats, H1, H1h);
  k5_g2     <<<64, 256, 0, stream>>>(H1h, W2ah, T, stats);
  k6_bn2    <<<1024, 256, 0, stream>>>(T, stats, T2h);
  k7_g3     <<<32, 256, 0, stream>>>(T2h, W2bh, H1, Zp, stats);
  k8_bn3    <<<256, 256, 0, stream>>>(Zp, stats, out);
}
